// Florence2VisionChannelAttention_61881888800856
// MI455X (gfx1250) — matrix operand order repaired and checked by reading, not compiled, not run on hardware
//
#include <hip/hip_runtime.h>
#include <hip/hip_bf16.h>
#include <math.h>

typedef __attribute__((ext_vector_type(16))) __bf16 v16bf;
typedef __attribute__((ext_vector_type(8)))  __bf16 v8bf;
typedef __attribute__((ext_vector_type(8)))  float  v8f;
typedef __attribute__((ext_vector_type(4)))  float  v4f;
typedef __attribute__((ext_vector_type(4)))  unsigned int u32x4;
typedef __attribute__((ext_vector_type(8)))  int  i32x8;
typedef __attribute__((ext_vector_type(4)))  int  i32x4;

#define SHUF16(lo,hi) __builtin_shufflevector((lo),(hi),0,1,2,3,4,5,6,7,8,9,10,11,12,13,14,15)

#if defined(__has_builtin)
# if __has_builtin(__builtin_amdgcn_tensor_load_to_lds) && __has_builtin(__builtin_amdgcn_s_wait_tensorcnt)
#  define HAVE_TDM 1
# else
#  define HAVE_TDM 0
# endif
#else
# define HAVE_TDM 0
#endif

static __device__ __forceinline__ v8f wmma_bf16(v16bf a, v16bf b, v8f c) {
  // (neg_a, A, neg_b, B, c_mod, C, reuse_a, reuse_b)
  return __builtin_amdgcn_wmma_f32_16x16x32_bf16(false, a, false, b, (short)0, c, false, false);
}

// flat pointer -> LDS byte offset (ISA 10.2: low 32 bits of generic LDS addr are the offset)
static __device__ __forceinline__ unsigned int lds_off(const void* p) {
  return (unsigned int)(unsigned long long)p;
}

#if HAVE_TDM
// TDM: DMA a 2D bf16 tile [tile_h rows x 32 elems] (row stride = stride elems) into LDS,
// inserting 16B of padding after every 64B row -> padded LDS row stride of 40 halfs (LDA).
static __device__ __forceinline__ void tdm_load_tile(unsigned int ldsaddr, const void* gptr,
                                                     int tile_h, int stride) {
  unsigned long long ga = (unsigned long long)gptr;
  u32x4 g0;
  g0[0] = 1u;                                   // count=1, user descriptor
  g0[1] = ldsaddr;                              // lds_addr
  g0[2] = (unsigned int)ga;                     // global_addr[31:0]
  g0[3] = (unsigned int)(ga >> 32) | (2u << 30);// global_addr[56:32] | type=2
  i32x8 g1;
  g1[0] = (1 << 16) | (1 << 20) | (3 << 22) | (3 << 25); // data_size=2B, pad_en, 64B interval, 16B pad
  g1[1] = (32 << 16);                           // tensor_dim0 = 32 (tile == tensor, no OOB)
  g1[2] = (tile_h & 0xFFFF) << 16;              // tensor_dim1 lo16
  g1[3] = (tile_h >> 16) | (32 << 16);          // tensor_dim1 hi | tile_dim0 = 32
  g1[4] = tile_h;                               // tile_dim1 (tile_dim2 = 0 -> 2D)
  g1[5] = stride;                               // tensor_dim0_stride lo32 (elems)
  g1[6] = 0;
  g1[7] = 0;
  i32x4 z4 = {};
#if defined(__clang_major__) && __clang_major__ >= 23
  i32x8 z8 = {};
  __builtin_amdgcn_tensor_load_to_lds(g0, g1, z4, z4, z8, 0);
#else
  __builtin_amdgcn_tensor_load_to_lds(g0, g1, z4, z4, 0);
#endif
}
#endif

// ---------------- problem dims ----------------
#define BB    8
#define NTOK  4096
#define CCH   768
#define GG    16
#define DH    48
#define MROWS (BB*NTOK)      // 32768

// ---------------- fp32 -> bf16 convert (4 elems/thread) ----------------
__global__ __launch_bounds__(256)
void k_cvt(const float* __restrict__ in, __bf16* __restrict__ out, int n4) {
  int i = blockIdx.x * 256 + threadIdx.x;
  if (i >= n4) return;
  v4f v = *(const v4f*)(in + (size_t)i * 4);
  __bf16* o = out + (size_t)i * 4;
  o[0] = (__bf16)v[0]; o[1] = (__bf16)v[1]; o[2] = (__bf16)v[2]; o[3] = (__bf16)v[3];
}

// ---------------- fp32 [K,N] -> bf16 [N,K] transpose-convert ----------------
__global__ __launch_bounds__(256)
void k_cvt_t(const float* __restrict__ w, __bf16* __restrict__ wT, int K, int N) {
  int idx = blockIdx.x * 256 + threadIdx.x;
  if (idx >= N * K) return;
  int n = idx / K, k = idx - n * K;
  wT[idx] = (__bf16)w[(size_t)k * N + n];
}

// ---------------- bf16 GEMM: out[M,Nc] = A[M,K] * BT[Nc,K]^T + bias ----------------
// 256 threads = 8 waves; block tile 128x128; wave tile 64x32; K-step 32.
// Tiles staged by TDM (double-buffered) when available.
#define LDA 40   // padded LDS row stride in halfs (32 data + 8 pad; matches TDM pad descriptor)
template <typename OutT>
__global__ __launch_bounds__(256)
void k_gemm(const __bf16* __restrict__ A, const __bf16* __restrict__ BT,
            const float* __restrict__ bias, OutT* __restrict__ out,
            int K, int Nc, int scale_cols, float scale) {
  __shared__ __bf16 As[2][128 * LDA];
  __shared__ __bf16 Bs[2][128 * LDA];
  const int tid  = threadIdx.x;
  const int wave = tid >> 5, lane = tid & 31;
  const int half = lane >> 4, l16 = lane & 15;
  const int wm = wave >> 2, wn = wave & 3;           // 2x4 wave grid
  const size_t blockN = (size_t)blockIdx.x * 128;
  const size_t blockM = (size_t)blockIdx.y * 128;
  const __bf16* Abase = A  + blockM * (size_t)K;
  const __bf16* Bbase = BT + blockN * (size_t)K;

  v8f acc[4][2] = {};
  int cur = 0;

#if HAVE_TDM
  if (wave == 0) {                       // scalar-uniform branch; TDM ignores EXEC
    tdm_load_tile(lds_off(&As[0][0]), Abase, 128, K);
    tdm_load_tile(lds_off(&Bs[0][0]), Bbase, 128, K);
    __builtin_amdgcn_s_wait_tensorcnt(0);
  }
  __syncthreads();
#endif

  for (int k0 = 0; k0 < K; k0 += 32) {
#if HAVE_TDM
    if (wave == 0 && k0 + 32 < K) {      // prefetch next K-slab into the other buffer
      tdm_load_tile(lds_off(&As[cur ^ 1][0]), Abase + k0 + 32, 128, K);
      tdm_load_tile(lds_off(&Bs[cur ^ 1][0]), Bbase + k0 + 32, 128, K);
    }
#else
    // fallback: cooperative vector loads, single buffer
#pragma unroll
    for (int s = 0; s < 2; ++s) {
      int seg = tid + s * 256;
      int row = seg >> 2, cs = (seg & 3) * 8;
      *(v8bf*)&As[0][row * LDA + cs] = *(const v8bf*)&Abase[row * (size_t)K + k0 + cs];
      *(v8bf*)&Bs[0][row * LDA + cs] = *(const v8bf*)&Bbase[row * (size_t)K + k0 + cs];
    }
    __syncthreads();
#endif
    const __bf16* Asb = As[cur];
    const __bf16* Bsb = Bs[cur];

    v16bf af[4], bfv[2];
#pragma unroll
    for (int i = 0; i < 4; ++i) {   // A frag: lanes0-15 K{0..7,16..23}, lanes16-31 K{8..15,24..31}
      int r = wm * 64 + i * 16 + l16;
      v8bf lo = *(const v8bf*)&Asb[r * LDA + (half ? 8 : 0)];
      v8bf hi = *(const v8bf*)&Asb[r * LDA + 16 + (half ? 8 : 0)];
      af[i] = SHUF16(lo, hi);
    }
#pragma unroll
    for (int j = 0; j < 2; ++j) {   // B frag: lanes0-15 K{0..7,16..23}, lanes16-31 K{8..15,24..31}
      int r = wn * 32 + j * 16 + l16;
      v8bf lo = *(const v8bf*)&Bsb[r * LDA + (half ? 8 : 0)];
      v8bf hi = *(const v8bf*)&Bsb[r * LDA + 16 + (half ? 8 : 0)];
      bfv[j] = SHUF16(lo, hi);
    }
#pragma unroll
    for (int i = 0; i < 4; ++i)
#pragma unroll
      for (int j = 0; j < 2; ++j)
        acc[i][j] = wmma_bf16(af[i], bfv[j], acc[i][j]);
    __syncthreads();

#if HAVE_TDM
    if (k0 + 32 < K) {
      if (wave == 0) __builtin_amdgcn_s_wait_tensorcnt(0);  // next slab landed
      __syncthreads();                                      // release all waves onto it
      cur ^= 1;
    }
#endif
  }

  // epilogue: D layout — lane = output column, VGPR r = row (half*8 + r)
#pragma unroll
  for (int i = 0; i < 4; ++i) {
#pragma unroll
    for (int j = 0; j < 2; ++j) {
      int    col = (int)blockN + wn * 32 + j * 16 + l16;
      size_t mb  = blockM + wm * 64 + i * 16 + half * 8;
      float  bv  = bias[col];
      float  sc  = (col < scale_cols) ? scale : 1.0f;
#pragma unroll
      for (int r = 0; r < 8; ++r) {
        float v = (acc[i][j][r] + bv) * sc;
        out[(mb + r) * (size_t)Nc + col] = (OutT)v;
      }
    }
  }
}

// ---------------- channel-attention logits + softmax ----------------
// One block per (b,g). S[48,48] = q^T k over 4096 tokens; each 16x16 tile owned by one wave.
__global__ __launch_bounds__(256)
void k_attn(const __bf16* __restrict__ qkv, __bf16* __restrict__ attn) {
  __shared__ __bf16 qt[48 * 36];   // [ch][tok], padded stride 36
  __shared__ __bf16 kt[48 * 36];
  __shared__ float  S[48 * 48];
  const int tid  = threadIdx.x;
  const int wave = tid >> 5, lane = tid & 31;
  const int half = lane >> 4, l16 = lane & 15;
  const int bg = blockIdx.x;               // 0..127
  const int b  = bg >> 4, g = bg & 15;
  const size_t rowbase = (size_t)b * NTOK;
  const int qoff = g * DH;

  const int i0 = wave / 3, j0 = wave - i0 * 3;  // waves 0..7 own tiles 0..7; wave0 also tile 8
  v8f acc0 = {}, acc1 = {};

  for (int t0 = 0; t0 < NTOK; t0 += 32) {
    // load 32-token q/k tiles, transpose into [ch][tok]
    for (int seg = tid; seg < 384; seg += 256) {
      int isK = seg >= 192;
      int s2  = isK ? seg - 192 : seg;
      int tok = s2 & 31;
      int cb  = (s2 >> 5) * 8;
      const __bf16* p = qkv + (rowbase + t0 + tok) * (size_t)(3 * CCH) + qoff + (isK ? CCH : 0) + cb;
      v8bf v = *(const v8bf*)p;
      __bf16* dst = isK ? kt : qt;
#pragma unroll
      for (int e = 0; e < 8; ++e) dst[(cb + e) * 36 + tok] = v[e];
    }
    __syncthreads();
    {
      int ra = i0 * 16 + l16;
      v8bf alo = *(const v8bf*)&qt[ra * 36 + (half ? 8 : 0)];
      v8bf ahi = *(const v8bf*)&qt[ra * 36 + 16 + (half ? 8 : 0)];
      v16bf afr = SHUF16(alo, ahi);
      int rb = j0 * 16 + l16;
      v8bf blo = *(const v8bf*)&kt[rb * 36 + (half ? 8 : 0)];
      v8bf bhi = *(const v8bf*)&kt[rb * 36 + 16 + (half ? 8 : 0)];
      v16bf bfr = SHUF16(blo, bhi);
      acc0 = wmma_bf16(afr, bfr, acc0);
      if (wave == 0) {                       // scalar-uniform branch keeps EXEC all-1
        int r2 = 32 + l16;
        v8bf a2lo = *(const v8bf*)&qt[r2 * 36 + (half ? 8 : 0)];
        v8bf a2hi = *(const v8bf*)&qt[r2 * 36 + 16 + (half ? 8 : 0)];
        v8bf b2lo = *(const v8bf*)&kt[r2 * 36 + (half ? 8 : 0)];
        v8bf b2hi = *(const v8bf*)&kt[r2 * 36 + 16 + (half ? 8 : 0)];
        acc1 = wmma_bf16(SHUF16(a2lo, a2hi), SHUF16(b2lo, b2hi), acc1);
      }
    }
    __syncthreads();
  }
  {
    int col = j0 * 16 + l16;
    int rb  = i0 * 16 + half * 8;
#pragma unroll
    for (int r = 0; r < 8; ++r) S[(rb + r) * 48 + col] = acc0[r];
    if (wave == 0) {
      int col2 = 32 + l16, rb2 = 32 + half * 8;
#pragma unroll
      for (int r = 0; r < 8; ++r) S[(rb2 + r) * 48 + col2] = acc1[r];
    }
  }
  __syncthreads();
  // row softmax -> attn[bg][48][64] (zero-padded K for the AV GEMM)
  if (tid < 48) {
    float mx = -3.4e38f;
    for (int e = 0; e < 48; ++e) mx = fmaxf(mx, S[tid * 48 + e]);
    float sum = 0.f;
    for (int e = 0; e < 48; ++e) sum += __expf(S[tid * 48 + e] - mx);
    float inv = 1.0f / sum;
    __bf16* row = attn + ((size_t)bg * 48 + tid) * 64;
    for (int e = 0; e < 48; ++e) row[e] = (__bf16)(__expf(S[tid * 48 + e] - mx) * inv);
    for (int e = 48; e < 64; ++e) row[e] = (__bf16)0.0f;
  }
}

// ---------------- out[b,n,g*48+dd] = sum_e attn[bg][dd][e] * v[b,n,g*48+e] ----------------
__global__ __launch_bounds__(256)
void k_av(const __bf16* __restrict__ qkv, const __bf16* __restrict__ attn,
          __bf16* __restrict__ aout) {
  const int tid  = threadIdx.x;
  const int wave = tid >> 5, lane = tid & 31;
  const int half = lane >> 4, l16 = lane & 15;
  const int bg = blockIdx.x;
  const int b  = bg >> 4, g = bg & 15;
  const int tbase = blockIdx.y * 128 + wave * 16;   // 16 tokens per wave

  // B fragments: attn rows [dd][64] (K = e, padded)
  v16bf bfv[2][3];
#pragma unroll
  for (int j = 0; j < 3; ++j)
#pragma unroll
    for (int s = 0; s < 2; ++s) {
      const __bf16* p = attn + ((size_t)bg * 48 + j * 16 + l16) * 64 + s * 32 + half * 8;
      v8bf lo = *(const v8bf*)p;
      v8bf hi = *(const v8bf*)(p + 16);
      bfv[s][j] = SHUF16(lo, hi);
    }
  // A fragments: v rows straight from global (48 valid channels, zero-pad to 64)
  const __bf16* vb = qkv + ((size_t)b * NTOK + tbase + l16) * (size_t)(3 * CCH) + 2 * CCH + g * DH;
  v8bf z8 = {};
  v8bf lo0 = *(const v8bf*)(vb + half * 8);
  v8bf hi0 = *(const v8bf*)(vb + 16 + half * 8);
  v16bf af0 = SHUF16(lo0, hi0);
  v8bf lo1 = *(const v8bf*)(vb + 32 + half * 8);   // e 32..47 valid
  v16bf af1 = SHUF16(lo1, z8);                     // e 48..63 zero

  v8f acc[3] = {};
#pragma unroll
  for (int j = 0; j < 3; ++j) {
    acc[j] = wmma_bf16(af0, bfv[0][j], acc[j]);
    acc[j] = wmma_bf16(af1, bfv[1][j], acc[j]);
  }
#pragma unroll
  for (int j = 0; j < 3; ++j) {
    int    col  = g * DH + j * 16 + l16;
    size_t mrow = (size_t)b * NTOK + tbase + half * 8;
#pragma unroll
    for (int r = 0; r < 8; ++r)
      aout[(mrow + r) * (size_t)CCH + col] = (__bf16)acc[j][r];
  }
}

// ---------------- host ----------------
extern "C" void kernel_launch(void* const* d_in, const int* in_sizes, int n_in,
                              void* d_out, int out_size, void* d_ws, size_t ws_size,
                              hipStream_t stream) {
  (void)in_sizes; (void)n_in; (void)out_size; (void)ws_size;
  const float* x     = (const float*)d_in[0];
  const float* w_qkv = (const float*)d_in[1];
  const float* b_qkv = (const float*)d_in[2];
  const float* w_prj = (const float*)d_in[3];
  const float* b_prj = (const float*)d_in[4];
  float* out = (float*)d_out;

  char* ws = (char*)d_ws;
  size_t off = 0;
  auto alloc = [&](size_t bytes) { size_t o = off; off = (off + bytes + 255) & ~(size_t)255; return o; };
  __bf16* x_bf  = (__bf16*)(ws + alloc((size_t)MROWS * CCH * 2));       // 50 MB
  __bf16* wqkvT = (__bf16*)(ws + alloc((size_t)3 * CCH * CCH * 2));     // 3.5 MB
  __bf16* wprjT = (__bf16*)(ws + alloc((size_t)CCH * CCH * 2));         // 1.2 MB
  __bf16* qkv   = (__bf16*)(ws + alloc((size_t)MROWS * 3 * CCH * 2));   // 151 MB
  __bf16* attn  = (__bf16*)(ws + alloc((size_t)BB * GG * 48 * 64 * 2)); // 0.8 MB
  __bf16* aout  = (__bf16*)(ws + alloc((size_t)MROWS * CCH * 2));       // 50 MB

  // 1) precision conversion / weight transposition
  k_cvt<<<(MROWS * CCH / 4 + 255) / 256, 256, 0, stream>>>(x, x_bf, MROWS * CCH / 4);
  k_cvt_t<<<(3 * CCH * CCH + 255) / 256, 256, 0, stream>>>(w_qkv, wqkvT, CCH, 3 * CCH);
  k_cvt_t<<<(CCH * CCH + 255) / 256, 256, 0, stream>>>(w_prj, wprjT, CCH, CCH);

  // 2) qkv = x @ w_qkv + b, q-columns (first 768) scaled by N^-1/2 = 1/64
  k_gemm<__bf16><<<dim3(3 * CCH / 128, MROWS / 128), 256, 0, stream>>>(
      x_bf, wqkvT, b_qkv, qkv, CCH, 3 * CCH, CCH, 0.015625f);

  // 3) channel-attention logits + softmax per (b,g)
  k_attn<<<BB * GG, 256, 0, stream>>>(qkv, attn);

  // 4) apply attention to v
  k_av<<<dim3(BB * GG, NTOK / 128), 256, 0, stream>>>(qkv, attn, aout);

  // 5) final projection (fp32 out)
  k_gemm<float><<<dim3(CCH / 128, MROWS / 128), 256, 0, stream>>>(
      aout, wprjT, b_prj, out, CCH, CCH, 0, 1.0f);
}
